// TriadTransformer_80092550136179
// MI455X (gfx1250) — compile-verified, matrix operand order checked by reading, not run on hardware
//
#include <hip/hip_runtime.h>
#include <cmath>

// ---------------------------------------------------------------------------
// CDNA5 (gfx1250) wave32 WMMA transformer pipeline.
// All matmuls use v_wmma_f32_16x16x32_bf16 (bf16 A/B, f32 accum).
// ---------------------------------------------------------------------------

typedef __bf16 v16bf __attribute__((ext_vector_type(16)));
typedef float  v8f   __attribute__((ext_vector_type(8)));

#define WMMA_BF16(a, b, c) \
  __builtin_amdgcn_wmma_f32_16x16x32_bf16(false, (a), false, (b), (short)0, (c), false, false)

#define NBINS 2048

__device__ __forceinline__ unsigned short f2bf(float f) {
  unsigned u = __float_as_uint(f);
  u += 0x7FFFu + ((u >> 16) & 1u);   // round-to-nearest-even
  return (unsigned short)(u >> 16);
}

// Fragment loader for 16-bit A/B operands of v_wmma_*_16x16x32.
// Per ISA layout: lane holds row (A) / column (B) = lane%16; element j holds
// k = j + (j>=8 ? 8 : 0) + (lane>=16 ? 8 : 0)  -> two contiguous 8-elem runs.
__device__ __forceinline__ v16bf load_frag(const unsigned short* p, int lh) {
  union { v16bf v; uint4 u[2]; } r;
  r.u[0] = *(const uint4*)(p + lh);
  r.u[1] = *(const uint4*)(p + 16 + lh);
  return r.v;
}

__device__ __forceinline__ v8f zero_acc() {
  v8f z;
#pragma unroll
  for (int i = 0; i < 8; ++i) z[i] = 0.0f;
  return z;
}

// ---------------------------------------------------------------------------
// fp32 -> bf16 conversion
// ---------------------------------------------------------------------------
__global__ void k_cvt_bf16(const float* __restrict__ in, unsigned short* __restrict__ out, int n) {
  int i = blockIdx.x * blockDim.x + threadIdx.x;
  if (i < n) out[i] = f2bf(in[i]);
}

// ---------------------------------------------------------------------------
// x = emb[src] * sqrt(D) + pos   (fp32 + bf16 copies)
// ---------------------------------------------------------------------------
__global__ void k_embed(const int* __restrict__ src, const float* __restrict__ emb,
                        const float* __restrict__ pos, float* __restrict__ xf,
                        unsigned short* __restrict__ xb, int n) {
  int i = blockIdx.x * blockDim.x + threadIdx.x;
  if (i >= n) return;
  int d = i & 511;
  int s = (i >> 9) & 511;
  int b = i >> 18;
  int tok = src[b * 512 + s];
  float v = emb[(size_t)tok * 512 + d] * 22.62741699796952f + pos[s * 512 + d];
  xf[i] = v;
  xb[i] = f2bf(v);
}

// ---------------------------------------------------------------------------
// C[M,N] = A[M,K] @ W[N,K]^T + bias, optional ReLU; fp32 and/or bf16 outputs.
// Block = 8 waves (256 thr); block tile 128(M) x 128(N); wave tile 32 x 64
// (two A fragments share four B fragments -> 8 WMMA per 12 b128 loads).
// K must be a multiple of 32; M a multiple of 128; N edge-guarded.
// ---------------------------------------------------------------------------
__global__ __launch_bounds__(256) void k_gemm(const unsigned short* __restrict__ A,
                                              const unsigned short* __restrict__ W,
                                              const float* __restrict__ bias,
                                              float* __restrict__ outF,
                                              unsigned short* __restrict__ outB,
                                              int M, int N, int K, int relu) {
  const int lane = threadIdx.x & 31;
  const int wave = threadIdx.x >> 5;
  const int l16  = lane & 15;
  const int lh   = (lane < 16) ? 0 : 8;
  const int m0   = blockIdx.y * 128 + (wave & 3) * 32;
  const int n0   = blockIdx.x * 128 + (wave >> 2) * 64;

  const unsigned short* pa0 = A + (size_t)(m0 + l16) * K;
  const unsigned short* pa1 = A + (size_t)(m0 + 16 + l16) * K;
  const unsigned short* pw[4];
  bool nv[4];
#pragma unroll
  for (int t = 0; t < 4; ++t) {
    int n = n0 + t * 16 + l16;
    nv[t] = (n < N);
    pw[t] = W + (size_t)(nv[t] ? n : 0) * K;
  }

  v8f acc0[4], acc1[4];
#pragma unroll
  for (int t = 0; t < 4; ++t) { acc0[t] = zero_acc(); acc1[t] = zero_acc(); }

  for (int kb = 0; kb < K; kb += 32) {
    __builtin_prefetch(pa0 + kb + 256, 0, 1);
    __builtin_prefetch(pa1 + kb + 256, 0, 1);
    v16bf a0 = load_frag(pa0 + kb, lh);
    v16bf a1 = load_frag(pa1 + kb, lh);
#pragma unroll
    for (int t = 0; t < 4; ++t) {
      v16bf b = load_frag(pw[t] + kb, lh);
      acc0[t] = WMMA_BF16(a0, b, acc0[t]);
      acc1[t] = WMMA_BF16(a1, b, acc1[t]);
    }
  }

  const int mrow = m0 + ((lane < 16) ? 0 : 8);
#pragma unroll
  for (int t = 0; t < 4; ++t) {
    int n = n0 + t * 16 + l16;
    if (n >= N) continue;
    float bv = bias ? bias[n] : 0.0f;
#pragma unroll
    for (int v = 0; v < 8; ++v) {
      float v0 = acc0[t][v] + bv;
      float v1 = acc1[t][v] + bv;
      if (relu) { v0 = fmaxf(v0, 0.0f); v1 = fmaxf(v1, 0.0f); }
      size_t i0 = (size_t)(mrow + v) * N + n;
      size_t i1 = (size_t)(mrow + 16 + v) * N + n;
      if (outF) { outF[i0] = v0; outF[i1] = v1; }
      if (outB) { outB[i0] = f2bf(v0); outB[i1] = f2bf(v1); }
    }
  }
}

// ---------------------------------------------------------------------------
// Fused attention: one wave32 workgroup per (b, h, 16-query tile).
// qkv layout [B,S,3*D] bf16; ctx layout [B,S,D] bf16.
// Phase 1: S = (Q K^T)/8 via WMMA into LDS.
// Phase 2: exact softmax per row (2 lanes per row + shfl_xor combine).
// Phase 3: ctx = P V via WMMA, V transposed per 32-key block into LDS.
// ---------------------------------------------------------------------------
__global__ __launch_bounds__(32) void k_attn(const unsigned short* __restrict__ qkv,
                                             unsigned short* __restrict__ ctx) {
  __shared__ __align__(16) float          sc[16 * 512];
  __shared__ __align__(16) unsigned short pb[16 * 512];
  __shared__ __align__(16) unsigned short vt[64 * 32];
  __shared__ float rowsum[16];

  const int lane = threadIdx.x;
  const int id   = blockIdx.x;
  const int qt   = id & 31;         // S/16 = 32 query tiles
  const int h    = (id >> 5) & 7;   // H = 8
  const int b    = id >> 8;         // B = 16
  const int l16  = lane & 15;
  const int lh   = (lane < 16) ? 0 : 8;
  const int qrow = (lane < 16) ? 0 : 8;

  // Q fragments for this tile (HD = 64 -> 2 K-steps of 32)
  const unsigned short* qp = qkv + (size_t)(b * 512 + qt * 16 + l16) * 1536 + h * 64;
  v16bf qa0 = load_frag(qp, lh);
  v16bf qa1 = load_frag(qp + 32, lh);

  // ---- scores ----
  for (int j = 0; j < 32; ++j) {
    const unsigned short* kp = qkv + (size_t)(b * 512 + j * 16 + l16) * 1536 + 512 + h * 64;
    v16bf kb0 = load_frag(kp, lh);
    v16bf kb1 = load_frag(kp + 32, lh);
    v8f acc = zero_acc();
    acc = WMMA_BF16(qa0, kb0, acc);
    acc = WMMA_BF16(qa1, kb1, acc);
#pragma unroll
    for (int v = 0; v < 8; ++v)
      sc[(v + qrow) * 512 + j * 16 + l16] = acc[v] * 0.125f;  // 1/sqrt(64)
  }
  __syncthreads();

  // ---- softmax: row = lane&15, column half = lane>>4 (256 cols each) ----
  {
    const int half = lane >> 4;             // 0 or 1
    const int rbase = l16 * 512 + half * 256;
    float m = -3.4e38f;
    for (int k = 0; k < 256; ++k) m = fmaxf(m, sc[rbase + k]);
    m = fmaxf(m, __shfl_xor(m, 16, 32));    // combine halves
    float s = 0.0f;
    for (int k = 0; k < 256; ++k) {
      float e = __expf(sc[rbase + k] - m);
      s += e;
      pb[rbase + k] = f2bf(e);
    }
    s += __shfl_xor(s, 16, 32);
    if (lane < 16) rowsum[lane] = s;
  }
  __syncthreads();

  // ---- ctx = P @ V ----
  v8f c[4];
#pragma unroll
  for (int t = 0; t < 4; ++t) c[t] = zero_acc();

  for (int kb = 0; kb < 16; ++kb) {
    // stage V^T for this 32-key block: vt[d][k_local]
    const unsigned short* vsrc = qkv + (size_t)(b * 512 + kb * 32 + lane) * 1536 + 1024 + h * 64;
    for (int d = 0; d < 64; ++d) vt[d * 32 + lane] = vsrc[d];
    __syncthreads();

    v16bf pa = load_frag(pb + l16 * 512 + kb * 32, lh);
#pragma unroll
    for (int t = 0; t < 4; ++t) {
      v16bf vb = load_frag(vt + (t * 16 + l16) * 32, lh);
      c[t] = WMMA_BF16(pa, vb, c[t]);
    }
    __syncthreads();
  }

  unsigned short* op = ctx + (size_t)(b * 512 + qt * 16) * 512 + h * 64;
#pragma unroll
  for (int t = 0; t < 4; ++t)
#pragma unroll
    for (int v = 0; v < 8; ++v) {
      float val = c[t][v] / rowsum[v + qrow];
      op[(v + qrow) * 512 + t * 16 + l16] = f2bf(val);
    }
}

// ---------------------------------------------------------------------------
// x = LayerNorm(x + y) * g + b   (D = 512, one block per row)
// ---------------------------------------------------------------------------
__global__ __launch_bounds__(256) void k_add_ln(const float* __restrict__ x,
                                                const float* __restrict__ y,
                                                const float* __restrict__ g,
                                                const float* __restrict__ b,
                                                float* __restrict__ xo,
                                                unsigned short* __restrict__ xbo) {
  __shared__ float s1[256], s2[256];
  const int t = threadIdx.x;
  const size_t base = (size_t)blockIdx.x * 512;
  float a0 = x[base + t] + y[base + t];
  float a1 = x[base + t + 256] + y[base + t + 256];
  s1[t] = a0 + a1;
  s2[t] = a0 * a0 + a1 * a1;
  __syncthreads();
  for (int o = 128; o > 0; o >>= 1) {
    if (t < o) { s1[t] += s1[t + o]; s2[t] += s2[t + o]; }
    __syncthreads();
  }
  float mean = s1[0] * (1.0f / 512.0f);
  float var  = s2[0] * (1.0f / 512.0f) - mean * mean;
  float rs   = rsqrtf(var + 1e-5f);
  float o0 = (a0 - mean) * rs * g[t] + b[t];
  float o1 = (a1 - mean) * rs * g[t + 256] + b[t + 256];
  xo[base + t] = o0;       xo[base + t + 256] = o1;
  xbo[base + t] = f2bf(o0); xbo[base + t + 256] = f2bf(o1);
}

// ---------------------------------------------------------------------------
// triad resonance: sum of squares + abs-max reduction
// scal[0]=sumsq scal[1]=maxbits scal[2]=scale_g scal[3]=binw scal[4]=thr
// ---------------------------------------------------------------------------
__global__ __launch_bounds__(256) void k_reduce(const float* __restrict__ x, int n,
                                                float* __restrict__ scal) {
  __shared__ float ss[256], sm[256];
  float lss = 0.0f, lmx = 0.0f;
  for (int i = blockIdx.x * blockDim.x + threadIdx.x; i < n; i += gridDim.x * blockDim.x) {
    float v = x[i];
    lss += v * v;
    lmx = fmaxf(lmx, fabsf(v));
  }
  ss[threadIdx.x] = lss;
  sm[threadIdx.x] = lmx;
  __syncthreads();
  for (int o = 128; o > 0; o >>= 1) {
    if (threadIdx.x < o) {
      ss[threadIdx.x] += ss[threadIdx.x + o];
      sm[threadIdx.x] = fmaxf(sm[threadIdx.x], sm[threadIdx.x + o]);
    }
    __syncthreads();
  }
  if (threadIdx.x == 0) {
    atomicAdd(&scal[0], ss[0]);
    atomicMax((unsigned*)&scal[1], __float_as_uint(sm[0]));  // |x| >= 0: uint order ok
  }
}

__global__ void k_prep(float* scal) {
  float norm = sqrtf(scal[0]);
  float maxa = __uint_as_float(((const unsigned*)scal)[1]);
  scal[2] = 2048.0f / (norm + 1e-8f);               // sqrt(B*S*D) / ||h||
  scal[3] = fmaxf(maxa, 1e-20f) * (1.0f / NBINS);   // bin width over |x|
}

__global__ __launch_bounds__(256) void k_hist(const float* __restrict__ x, int n,
                                              const float* __restrict__ scal,
                                              unsigned* __restrict__ hist) {
  __shared__ unsigned h[NBINS];
  for (int i = threadIdx.x; i < NBINS; i += 256) h[i] = 0;
  __syncthreads();
  float inv = 1.0f / scal[3];
  for (int i = blockIdx.x * blockDim.x + threadIdx.x; i < n; i += gridDim.x * blockDim.x) {
    int bin = (int)(fabsf(x[i]) * inv);
    bin = min(bin, NBINS - 1);
    atomicAdd(&h[bin], 1u);
  }
  __syncthreads();
  for (int i = threadIdx.x; i < NBINS; i += 256)
    if (h[i]) atomicAdd(&hist[i], h[i]);
}

__global__ void k_quantile(const unsigned* __restrict__ hist, float* scal, int n) {
  double rank = 0.75 * (double)(n - 1);
  double cum = 0.0;
  float binw = scal[3];
  float thr = (float)NBINS * binw;
  for (int i = 0; i < NBINS; ++i) {
    double c = (double)hist[i];
    if (cum + c > rank) {
      double frac = (c > 0.0) ? (rank - cum) / c : 0.0;
      thr = (float)(((double)i + frac) * (double)binw);
      break;
    }
    cum += c;
  }
  scal[4] = thr;  // threshold in |x| units (quantile commutes with positive scale)
}

__global__ void k_threshold(float* __restrict__ x, unsigned short* __restrict__ xb,
                            const float* __restrict__ scal, float damp, int n) {
  int i = blockIdx.x * blockDim.x + threadIdx.x;
  if (i >= n) return;
  float v = x[i];
  float r = (fabsf(v) > scal[4]) ? v * scal[2] * damp : 0.0f;
  x[i] = r;
  xb[i] = f2bf(r);
}

// ---------------------------------------------------------------------------
// host launcher
// ---------------------------------------------------------------------------
extern "C" void kernel_launch(void* const* d_in, const int* in_sizes, int n_in,
                              void* d_out, int out_size, void* d_ws, size_t ws_size,
                              hipStream_t stream) {
  (void)in_sizes; (void)n_in; (void)out_size; (void)ws_size;

  const int*   src   = (const int*)  d_in[0];
  const float* emb   = (const float*)d_in[1];
  const float* pos   = (const float*)d_in[2];
  const float* qkvw  = (const float*)d_in[3];
  const float* qkvb  = (const float*)d_in[4];
  const float* oww   = (const float*)d_in[5];
  const float* owb   = (const float*)d_in[6];
  const float* ln1s  = (const float*)d_in[7];
  const float* ln1b  = (const float*)d_in[8];
  const float* ln2s  = (const float*)d_in[9];
  const float* ln2b  = (const float*)d_in[10];
  const float* f1w   = (const float*)d_in[11];
  const float* f1b   = (const float*)d_in[12];
  const float* f2w   = (const float*)d_in[13];
  const float* f2bb  = (const float*)d_in[14];
  const float* hw    = (const float*)d_in[15];
  const float* hb    = (const float*)d_in[16];
  float* out = (float*)d_out;

  const int B = 16, S = 512, D = 512, L = 6, FF = 2048, V = 10000;
  const int M = B * S;              // 8192 tokens
  const size_t ND = (size_t)M * D;  // 4194304

  char* ws = (char*)d_ws;
  size_t off = 0;
  auto alloc = [&](size_t bytes) -> char* {
    char* p = ws + off;
    off = (off + bytes + 255) & ~(size_t)255;
    return p;
  };

  unsigned short* qkvw_b = (unsigned short*)alloc((size_t)L * 3 * D * D * 2);
  unsigned short* oww_b  = (unsigned short*)alloc((size_t)L * D * D * 2);
  unsigned short* f1w_b  = (unsigned short*)alloc((size_t)L * FF * D * 2);
  unsigned short* f2w_b  = (unsigned short*)alloc((size_t)L * D * FF * 2);
  unsigned short* hw_b   = (unsigned short*)alloc((size_t)V * D * 2);
  float*          xf     = (float*)alloc(ND * 4);
  unsigned short* xb     = (unsigned short*)alloc(ND * 2);
  unsigned short* qkvo   = (unsigned short*)alloc((size_t)M * 3 * D * 2);
  unsigned short* ctx    = (unsigned short*)alloc(ND * 2);
  float*          yf     = (float*)alloc(ND * 4);
  unsigned short* h1     = (unsigned short*)alloc((size_t)M * FF * 2);
  float*          scal   = (float*)alloc(256 + NBINS * 4);
  unsigned*       hist   = (unsigned*)(scal + 64);

  auto cvt = [&](const float* in, unsigned short* o, size_t n) {
    k_cvt_bf16<<<dim3((unsigned)((n + 255) / 256)), 256, 0, stream>>>(in, o, (int)n);
  };
  cvt(qkvw, qkvw_b, (size_t)L * 3 * D * D);
  cvt(oww,  oww_b,  (size_t)L * D * D);
  cvt(f1w,  f1w_b,  (size_t)L * FF * D);
  cvt(f2w,  f2w_b,  (size_t)L * D * FF);
  cvt(hw,   hw_b,   (size_t)V * D);

  k_embed<<<dim3((unsigned)((ND + 255) / 256)), 256, 0, stream>>>(src, emb, pos, xf, xb, (int)ND);

  for (int i = 0; i < L; ++i) {
    // QKV projection -> bf16 [M, 1536]
    k_gemm<<<dim3(1536 / 128, M / 128), 256, 0, stream>>>(
        xb, qkvw_b + (size_t)i * 3 * D * D, qkvb + (size_t)i * 3 * D,
        nullptr, qkvo, M, 3 * D, D, 0);
    // fused attention -> ctx bf16 [M, D]
    k_attn<<<dim3(B * 8 * (S / 16)), 32, 0, stream>>>(qkvo, ctx);
    // output projection -> yf fp32
    k_gemm<<<dim3(512 / 128, M / 128), 256, 0, stream>>>(
        ctx, oww_b + (size_t)i * D * D, owb + (size_t)i * D,
        yf, nullptr, M, D, D, 0);
    // residual + LN1
    k_add_ln<<<dim3(M), 256, 0, stream>>>(xf, yf, ln1s + i * D, ln1b + i * D, xf, xb);
    // FF1 + ReLU -> h1 bf16 [M, FF]
    k_gemm<<<dim3(FF / 128, M / 128), 256, 0, stream>>>(
        xb, f1w_b + (size_t)i * FF * D, f1b + (size_t)i * FF,
        nullptr, h1, M, FF, D, 1);
    // FF2 -> yf fp32
    k_gemm<<<dim3(512 / 128, M / 128), 256, 0, stream>>>(
        h1, f2w_b + (size_t)i * D * FF, f2bb + (size_t)i * D,
        yf, nullptr, M, D, FF, 0);
    // residual + LN2
    k_add_ln<<<dim3(M), 256, 0, stream>>>(xf, yf, ln2s + i * D, ln2b + i * D, xf, xb);

    // triad resonance cycle
    hipMemsetAsync(scal, 0, 256 + NBINS * 4, stream);
    k_reduce<<<dim3(1024), 256, 0, stream>>>(xf, (int)ND, scal);
    k_prep<<<1, 1, 0, stream>>>(scal);
    k_hist<<<dim3(1024), 256, 0, stream>>>(xf, (int)ND, scal, hist);
    k_quantile<<<1, 1, 0, stream>>>(hist, scal, (int)ND);
    float damp = powf(0.99f, (float)(1 + i) / 6.0f);  // DAMP^(1/L) * DAMP^(i/L)
    k_threshold<<<dim3((unsigned)((ND + 255) / 256)), 256, 0, stream>>>(xf, xb, scal, damp, (int)ND);
  }

  // head: [M, V] fp32 to d_out (N edge-guarded: 10000 % 128 != 0)
  k_gemm<<<dim3((V + 127) / 128, M / 128), 256, 0, stream>>>(
      xb, hw_b, hb, out, nullptr, M, V, D, 0);
}
